// TreeSSMReadout_81415400063704
// MI455X (gfx1250) — matrix operand order repaired and checked by reading, not compiled, not run on hardware
//
#include <hip/hip_runtime.h>
#include <hip/hip_bf16.h>
#include <math.h>

// ---------------- problem constants ----------------
#define N_NODES   33825
#define D_INPUT   1537
#define D_SSM     256
#define D_STATE   16
#define LD_DBC    320            // delta(256) | B(16) | C(16) | pad(32)
#define KC1       49             // ceil(1537/32)
#define NT1       16             // 256/16
#define KC2       8              // 256/32
#define NT2       20             // 320/16
#define H_NODES   1057           // levels 0..2 keep h (parents of everything)

#define BLK_M     128
#define TILE_K    32
#define LDSW      36             // LDS row stride (f32) -> conflict-free, 16B aligned
#define BUFSZ     (BLK_M * LDSW) // 4608 f32 per buffer

typedef __attribute__((ext_vector_type(16))) __bf16 v16bf;
typedef __attribute__((ext_vector_type(8)))  float  v8f;
typedef __attribute__((ext_vector_type(4)))  unsigned v4u;
typedef __attribute__((ext_vector_type(8)))  int    v8i;
typedef __attribute__((ext_vector_type(4)))  int    v4i;

union BF16Frag { v16bf v; unsigned u[8]; unsigned short s[16]; };

__device__ inline unsigned short bf16_rne(float x) {
  unsigned u = __float_as_uint(x);
  unsigned r = u + 0x7FFFu + ((u >> 16) & 1u);
  return (unsigned short)(r >> 16);
}
__device__ inline float bf16_f32(unsigned short h) {
  return __uint_as_float(((unsigned)h) << 16);
}

#define WMMA_BF16(A_, B_, C_) \
  __builtin_amdgcn_wmma_f32_16x16x32_bf16(false, (A_), false, (B_), (short)0, (C_), false, false)

// ---------------- TDM: async 2D tile load global->LDS ----------------
// D# group0: count=1 | lds_addr | global_addr | type=2
// D# group1: data_size=4B, pad_enable, pad_interval=32dw, pad_amount=4dw (-> LDS stride 36),
//            tensor_dim0/1 = remaining K/M extent (TDM zero-fills OOB), tile 32x128,
//            tensor_dim0_stride = lda.
#if __has_include(<hip/amd_detail/amd_gfx1250_TDM.h>)
#define ATHENA_TDM_6ARG 1
#endif

__device__ inline void tdm_load_tile(unsigned ldsOff, const float* gptr,
                                     int dimK_rem, int dimM_rem, int lda) {
  unsigned long long ga = (unsigned long long)(uintptr_t)gptr;
  v4u g0;
  g0[0] = 1u;                                        // count=1 (valid), user mode
  g0[1] = ldsOff;                                    // LDS byte address
  g0[2] = (unsigned)(ga & 0xffffffffu);              // global_addr[31:0]
  g0[3] = (unsigned)((ga >> 32) & 0x01ffffffu) | (2u << 30);  // addr[56:32] | type=2
  v8i g1;
  unsigned dk = (unsigned)dimK_rem, dm = (unsigned)dimM_rem;
  g1[0] = (int)((2u << 16) | (1u << 20) | (4u << 22) | (3u << 25)); // 4B elems, pad 4dw/32dw
  g1[1] = (int)((dk & 0xffffu) << 16);               // tensor_dim0 lo16 @ [63:48]
  g1[2] = (int)(((dk >> 16) & 0xffffu) | ((dm & 0xffffu) << 16));   // dim0 hi | dim1 lo
  g1[3] = (int)(((dm >> 16) & 0xffffu) | ((unsigned)TILE_K << 16)); // dim1 hi | tile_dim0=32
  g1[4] = BLK_M;                                     // tile_dim1=128 | tile_dim2=0
  g1[5] = lda;                                       // tensor_dim0_stride lo32
  g1[6] = 0;                                         // stride hi | dim1_stride lo (unused, 2D)
  g1[7] = 0;
  v4i gz = {0, 0, 0, 0};
#ifdef ATHENA_TDM_6ARG
  v8i gz8 = {0, 0, 0, 0, 0, 0, 0, 0};
  __builtin_amdgcn_tensor_load_to_lds(g0, g1, gz, gz, gz8, 0);
#else
  __builtin_amdgcn_tensor_load_to_lds(g0, g1, gz, gz, 0);
#endif
}

// ---------------- weight packing (bf16 hi/lo fragments, contiguous-K B layout) ----------------
// B-frag (32x16 bf16): lane l<16 -> column N=l, K=0..15; lane l>=16 -> N=l-16, K=16..31.
__global__ __launch_bounds__(256) void pack_w1_kernel(
    const float* __restrict__ W, unsigned short* __restrict__ hi, unsigned short* __restrict__ lo) {
  int idx = blockIdx.x * 256 + threadIdx.x;
  if (idx >= KC1 * NT1 * 512) return;
  int frag = idx >> 9, e = idx & 511, lane = e >> 4, j = e & 15;
  int kc = frag / NT1, nt = frag % NT1;
  int k = kc * 32 + (lane >> 4) * 16 + j;
  int n = nt * 16 + (lane & 15);
  float w = (k < D_INPUT) ? W[(size_t)k * D_SSM + n] : 0.f;
  unsigned short h = bf16_rne(w);
  hi[idx] = h;
  lo[idx] = bf16_rne(w - bf16_f32(h));
}

__global__ __launch_bounds__(256) void pack_w2_kernel(
    const float* __restrict__ Wd, const float* __restrict__ WB, const float* __restrict__ WC,
    unsigned short* __restrict__ hi, unsigned short* __restrict__ lo) {
  int idx = blockIdx.x * 256 + threadIdx.x;
  if (idx >= KC2 * NT2 * 512) return;
  int frag = idx >> 9, e = idx & 511, lane = e >> 4, j = e & 15;
  int kc = frag / NT2, nt = frag % NT2;
  int k = kc * 32 + (lane >> 4) * 16 + j;           // K=256 exact
  int n = nt * 16 + (lane & 15);                    // 0..319
  float w = 0.f;
  if (n < 256)      w = Wd[(size_t)k * 256 + n];
  else if (n < 272) w = WB[(size_t)k * 16 + (n - 256)];
  else if (n < 288) w = WC[(size_t)k * 16 + (n - 272)];
  unsigned short h = bf16_rne(w);
  hi[idx] = h;
  lo[idx] = bf16_rne(w - bf16_f32(h));
}

__global__ __launch_bounds__(256) void prep_misc_kernel(
    const float* __restrict__ X, const float* __restrict__ A_log,
    const float* __restrict__ b_delta, const float* __restrict__ b_B, const float* __restrict__ b_C,
    float* __restrict__ logw, float* __restrict__ aneg, float* __restrict__ bias2) {
  int idx = blockIdx.x * 256 + threadIdx.x;
  if (idx < N_NODES) {
    logw[idx] = X[(size_t)idx * D_INPUT + (D_INPUT - 1)];
  } else if (idx < N_NODES + D_SSM * D_STATE) {
    int i = idx - N_NODES;
    aneg[i] = -__expf(A_log[i]);
  } else if (idx < N_NODES + D_SSM * D_STATE + LD_DBC) {
    int i = idx - N_NODES - D_SSM * D_STATE;
    float b = 0.f;
    if (i < 256)      b = b_delta[i];
    else if (i < 272) b = b_B[i - 256];
    else if (i < 288) b = b_C[i - 272];
    bias2[i] = b;
  }
}

// ---------------- bf16x3 WMMA GEMM, TDM-staged A, full-N block ----------------
// Block: 128 x (NT*16) output, 8 waves = 4(M, 32 rows) x 2(N, NTPW tiles).
// A streamed from HBM exactly once (grid.y == 1). TDM double-buffers A tiles in LDS.
// mode 0: identity. mode 1: cols<256 -> softplus(v)*sigmoid(logw[row]*Ww[col]+bw[col]).
template <int NTPW>
__global__ __launch_bounds__(256) void gemm_bf16x3_tdm_kernel(
    const float* __restrict__ A, int lda, int M, int K, int KC,
    const unsigned short* __restrict__ fragHi, const unsigned short* __restrict__ fragLo,
    const float* __restrict__ bias, float* __restrict__ out, int ldo,
    int mode, const float* __restrict__ logw, const float* __restrict__ Ww,
    const float* __restrict__ bw) {
  constexpr int NT = 2 * NTPW;
  __shared__ float As[2 * BUFSZ];
  const int tid = threadIdx.x;
  const int lane = tid & 31;
  const int wave = tid >> 5;
  const int wm = wave & 3;          // 0..3 (M, 32 rows each)
  const int wn = wave >> 2;         // 0..1 (N, NTPW tiles each)
  const int m0 = blockIdx.x * BLK_M;
  const int halfsel = lane >> 4;
  const int lr = lane & 15;
  const bool issuer = (tid < 32);   // wave 0 (wave-uniform predicate)

  v8f acc[2][NTPW];
  const v8f vzero = {0.f, 0.f, 0.f, 0.f, 0.f, 0.f, 0.f, 0.f};
  #pragma unroll
  for (int mt = 0; mt < 2; ++mt)
    #pragma unroll
    for (int nt = 0; nt < NTPW; ++nt) acc[mt][nt] = vzero;

  const int dimM_rem = M - m0;
  if (issuer) {
    tdm_load_tile((unsigned)(uintptr_t)&As[0], A + (size_t)m0 * lda, K, dimM_rem, lda);
  }

  for (int kc = 0; kc < KC; ++kc) {
    const int cur = kc & 1;
    if (issuer) __builtin_amdgcn_s_wait_tensorcnt(0);   // buf[cur] DMA complete
    __syncthreads();                                    // publish buf[cur]; buf[1-cur] free
    if (issuer && (kc + 1) < KC) {
      int k0n = (kc + 1) * TILE_K;
      tdm_load_tile((unsigned)(uintptr_t)&As[(1 - cur) * BUFSZ],
                    A + (size_t)m0 * lda + k0n, K - k0n, dimM_rem, lda);
    }
    const float* Abuf = &As[cur * BUFSZ];

    // A fragments (16-bit A 16x32 layout), hi/lo bf16 split
    BF16Frag aHi[2], aLo[2];
    #pragma unroll
    for (int mt = 0; mt < 2; ++mt) {
      const float* src = &Abuf[(wm * 32 + mt * 16 + lr) * LDSW];
      int kb = halfsel * 16;
      #pragma unroll
      for (int j = 0; j < 8; ++j) {
        float x0 = src[kb + j];           // K = halfsel*16 + j
        float x1 = src[kb + 8 + j];       // K = halfsel*16 + 8 + j
        unsigned short h0 = bf16_rne(x0);
        unsigned short h1 = bf16_rne(x1);
        aHi[mt].s[j]     = h0;
        aHi[mt].s[8 + j] = h1;
        aLo[mt].s[j]     = bf16_rne(x0 - bf16_f32(h0));
        aLo[mt].s[8 + j] = bf16_rne(x1 - bf16_f32(h1));
      }
    }

    #pragma unroll
    for (int nt = 0; nt < NTPW; ++nt) {
      int gnt = wn * NTPW + nt;
      size_t base = ((size_t)kc * NT + gnt) * 512 + (size_t)lane * 16;
      BF16Frag bHi, bLo;
      const uint4* ph = reinterpret_cast<const uint4*>(fragHi + base);
      const uint4* pl = reinterpret_cast<const uint4*>(fragLo + base);
      uint4 h0 = ph[0], h1 = ph[1];
      uint4 l0 = pl[0], l1 = pl[1];
      bHi.u[0] = h0.x; bHi.u[1] = h0.y; bHi.u[2] = h0.z; bHi.u[3] = h0.w;
      bHi.u[4] = h1.x; bHi.u[5] = h1.y; bHi.u[6] = h1.z; bHi.u[7] = h1.w;
      bLo.u[0] = l0.x; bLo.u[1] = l0.y; bLo.u[2] = l0.z; bLo.u[3] = l0.w;
      bLo.u[4] = l1.x; bLo.u[5] = l1.y; bLo.u[6] = l1.z; bLo.u[7] = l1.w;
      #pragma unroll
      for (int mt = 0; mt < 2; ++mt) {
        acc[mt][nt] = WMMA_BF16(aHi[mt].v, bHi.v, acc[mt][nt]);  // hi*hi
        acc[mt][nt] = WMMA_BF16(aHi[mt].v, bLo.v, acc[mt][nt]);  // hi*lo
        acc[mt][nt] = WMMA_BF16(aLo[mt].v, bHi.v, acc[mt][nt]);  // lo*hi
      }
    }
  }

  // epilogue + store (C/D layout: elem e -> M = e + halfsel*8, N = lane%16)
  #pragma unroll
  for (int mt = 0; mt < 2; ++mt) {
    #pragma unroll
    for (int nt = 0; nt < NTPW; ++nt) {
      int col = (wn * NTPW + nt) * 16 + lr;
      #pragma unroll
      for (int e = 0; e < 8; ++e) {
        int row = m0 + wm * 32 + mt * 16 + halfsel * 8 + e;
        if (row < M) {
          float v = acc[mt][nt][e] + bias[col];
          if (mode == 1 && col < 256) {
            float g  = 1.f / (1.f + __expf(-(logw[row] * Ww[col] + bw[col])));
            float sp = fmaxf(v, 0.f) + log1pf(__expf(-fabsf(v)));
            v = sp * g;
          }
          out[(size_t)row * ldo + col] = v;
        }
      }
    }
  }
}

// ---------------- scan: levels 0..2 (h stored per node) ----------------
__global__ __launch_bounds__(256) void scan_small_kernel(
    const float* __restrict__ dbc, const float* __restrict__ xp, const float* __restrict__ aneg,
    float* __restrict__ hbuf, const float* __restrict__ Dv,
    const float* __restrict__ gamma, const float* __restrict__ beta,
    float* __restrict__ out, int lo, int isRoot) {
  __shared__ float bcS[32];
  __shared__ float red[256];
  __shared__ float red2[256];
  int n = lo + blockIdx.x;
  int tid = threadIdx.x;                    // = d
  float delta = dbc[(size_t)n * LD_DBC + tid];
  float xpv   = xp[(size_t)n * D_SSM + tid];
  if (tid < 32) bcS[tid] = dbc[(size_t)n * LD_DBC + 256 + tid];
  __syncthreads();
  float du = delta * xpv;
  float h[16];
  if (isRoot) {
    #pragma unroll
    for (int s = 0; s < 16; ++s) h[s] = du * bcS[s];
  } else {
    int p = (n - 1) >> 5;
    #pragma unroll
    for (int s = 0; s < 16; ++s)
      h[s] = __expf(delta * aneg[tid * 16 + s]) * hbuf[(size_t)p * 4096 + tid * 16 + s] + du * bcS[s];
  }
  float y = 0.f;
  #pragma unroll
  for (int s = 0; s < 16; ++s) {
    hbuf[(size_t)n * 4096 + tid * 16 + s] = h[s];
    y += h[s] * bcS[16 + s];
  }
  y += Dv[tid] * xpv;
  red[tid] = y; red2[tid] = y * y;
  __syncthreads();
  for (int off = 128; off > 0; off >>= 1) {
    if (tid < off) { red[tid] += red[tid + off]; red2[tid] += red2[tid + off]; }
    __syncthreads();
  }
  float mu  = red[0] * (1.f / 256.f);
  float var = red2[0] * (1.f / 256.f) - mu * mu;
  out[(size_t)n * D_SSM + tid] = (y - mu) * rsqrtf(var + 1e-5f) * gamma[tid] + beta[tid];
}

// ---------------- level 3 fused: h never materialized (saves ~1 GB HBM traffic) ----------------
__global__ __launch_bounds__(256) void level3_kernel(
    const float* __restrict__ dbc, const float* __restrict__ xp, const float* __restrict__ aneg,
    const float* __restrict__ hbuf, const float* __restrict__ Dv,
    const float* __restrict__ gamma, const float* __restrict__ beta, float* __restrict__ out) {
  __shared__ float hparS[256 * 17];   // stride 17: conflict-free d*17+s access
  __shared__ float aS[256 * 17];
  __shared__ float bcS[32];
  __shared__ float red[256];
  __shared__ float red2[256];
  int p = 33 + blockIdx.x;            // level-2 parent node id
  int tid = threadIdx.x;              // = d
  for (int i = tid; i < 4096; i += 256) {
    int d = i >> 4, s = i & 15;
    hparS[d * 17 + s] = hbuf[(size_t)p * 4096 + i];
    aS[d * 17 + s]    = aneg[i];
  }
  __syncthreads();
  int firstChild = p * 32 + 1;
  float Dd = Dv[tid], gm = gamma[tid], bt = beta[tid];
  const float* ap = &aS[tid * 17];
  const float* hp = &hparS[tid * 17];
  for (int c = 0; c < 32; ++c) {
    int n = firstChild + c;
    float delta = dbc[(size_t)n * LD_DBC + tid];
    float xpv   = xp[(size_t)n * D_SSM + tid];
    if (tid < 32) bcS[tid] = dbc[(size_t)n * LD_DBC + 256 + tid];
    __syncthreads();
    float du = delta * xpv;
    float y = 0.f;
    #pragma unroll
    for (int s = 0; s < 16; ++s) {
      float h = __expf(delta * ap[s]) * hp[s] + du * bcS[s];
      y += h * bcS[16 + s];
    }
    y += Dd * xpv;
    red[tid] = y; red2[tid] = y * y;
    __syncthreads();
    for (int off = 128; off > 0; off >>= 1) {
      if (tid < off) { red[tid] += red[tid + off]; red2[tid] += red2[tid + off]; }
      __syncthreads();
    }
    float mu  = red[0] * (1.f / 256.f);
    float var = red2[0] * (1.f / 256.f) - mu * mu;
    out[(size_t)n * D_SSM + tid] = (y - mu) * rsqrtf(var + 1e-5f) * gm + bt;
    __syncthreads();                  // bcS/red reused next child
  }
}

// ---------------- host launch ----------------
static inline size_t alignup(size_t x) { return (x + 255) & ~(size_t)255; }

extern "C" void kernel_launch(void* const* d_in, const int* in_sizes, int n_in,
                              void* d_out, int out_size, void* d_ws, size_t ws_size,
                              hipStream_t stream) {
  (void)in_sizes; (void)n_in; (void)out_size; (void)ws_size;
  const float* X       = (const float*)d_in[0];
  const float* W_in    = (const float*)d_in[2];
  const float* b_in    = (const float*)d_in[3];
  const float* W_delta = (const float*)d_in[4];
  const float* b_delta = (const float*)d_in[5];
  const float* W_w     = (const float*)d_in[6];
  const float* b_w     = (const float*)d_in[7];
  const float* A_log   = (const float*)d_in[8];
  const float* Dvec    = (const float*)d_in[9];
  const float* W_B     = (const float*)d_in[10];
  const float* b_B     = (const float*)d_in[11];
  const float* W_C     = (const float*)d_in[12];
  const float* b_C     = (const float*)d_in[13];
  const float* gamma   = (const float*)d_in[14];
  const float* beta    = (const float*)d_in[15];
  float* out = (float*)d_out;

  char* ws = (char*)d_ws;
  size_t o = 0;
  float* ws_xp   = (float*)(ws + o); o += alignup((size_t)N_NODES * D_SSM * 4);
  float* ws_dbc  = (float*)(ws + o); o += alignup((size_t)N_NODES * LD_DBC * 4);
  float* ws_h    = (float*)(ws + o); o += alignup((size_t)H_NODES * 4096 * 4);
  unsigned short* ws_w1h = (unsigned short*)(ws + o); o += alignup((size_t)KC1 * NT1 * 512 * 2);
  unsigned short* ws_w1l = (unsigned short*)(ws + o); o += alignup((size_t)KC1 * NT1 * 512 * 2);
  unsigned short* ws_w2h = (unsigned short*)(ws + o); o += alignup((size_t)KC2 * NT2 * 512 * 2);
  unsigned short* ws_w2l = (unsigned short*)(ws + o); o += alignup((size_t)KC2 * NT2 * 512 * 2);
  float* ws_logw  = (float*)(ws + o); o += alignup((size_t)N_NODES * 4);
  float* ws_aneg  = (float*)(ws + o); o += alignup((size_t)D_SSM * D_STATE * 4);
  float* ws_bias2 = (float*)(ws + o); o += alignup((size_t)LD_DBC * 4);

  // prep
  pack_w1_kernel<<<(KC1 * NT1 * 512 + 255) / 256, 256, 0, stream>>>(W_in, ws_w1h, ws_w1l);
  pack_w2_kernel<<<(KC2 * NT2 * 512 + 255) / 256, 256, 0, stream>>>(W_delta, W_B, W_C, ws_w2h, ws_w2l);
  int miscTot = N_NODES + D_SSM * D_STATE + LD_DBC;
  prep_misc_kernel<<<(miscTot + 255) / 256, 256, 0, stream>>>(X, A_log, b_delta, b_B, b_C,
                                                              ws_logw, ws_aneg, ws_bias2);

  // GEMM 1: X_p = X @ W_in + b_in   (full N=256 per block; X streamed once)
  dim3 g1((N_NODES + BLK_M - 1) / BLK_M, 1);
  gemm_bf16x3_tdm_kernel<8><<<g1, 256, 0, stream>>>(X, D_INPUT, N_NODES, D_INPUT, KC1,
                                                    ws_w1h, ws_w1l, b_in, ws_xp, D_SSM,
                                                    0, nullptr, nullptr, nullptr);

  // GEMM 2: [delta | B | C] = act(X_p @ [W_delta|W_B|W_C] + bias2)  (full N=320 per block)
  gemm_bf16x3_tdm_kernel<10><<<g1, 256, 0, stream>>>(ws_xp, D_SSM, N_NODES, D_SSM, KC2,
                                                     ws_w2h, ws_w2l, ws_bias2, ws_dbc, LD_DBC,
                                                     1, ws_logw, W_w, b_w);

  // tree scan levels 0..2 (h kept), level 3 fused readout (h never written)
  scan_small_kernel<<<1, 256, 0, stream>>>(ws_dbc, ws_xp, ws_aneg, ws_h, Dvec, gamma, beta, out, 0, 1);
  scan_small_kernel<<<32, 256, 0, stream>>>(ws_dbc, ws_xp, ws_aneg, ws_h, Dvec, gamma, beta, out, 1, 0);
  scan_small_kernel<<<1024, 256, 0, stream>>>(ws_dbc, ws_xp, ws_aneg, ws_h, Dvec, gamma, beta, out, 33, 0);
  level3_kernel<<<1024, 256, 0, stream>>>(ws_dbc, ws_xp, ws_aneg, ws_h, Dvec, gamma, beta, out);
}
